// DynamicRVQ_87265145520600
// MI455X (gfx1250) — matrix operand order repaired and checked by reading, not compiled, not run on hardware
//
#include <hip/hip_runtime.h>
#include <hip/hip_bf16.h>

// ---------------------------------------------------------------------------
// Dynamic RVQ forward for MI455X (gfx1250, wave32, WMMA).
//   B=16, D=256, T=2048, K=1024, N_CB=8, N = B*T = 32768 rows.
// Distance argmin per codebook via v_wmma_f32_16x16x32_bf16 with a bf16
// hi/lo split of the residual (≈fp32-accurate dot products).
// Codebook chunks staged LDS-side with double-buffered
// global_load_async_to_lds_b128 (ASYNCcnt), overlapped with WMMA compute.
// quantized_total = x - residual_final  (exact identity of the reference).
// ---------------------------------------------------------------------------

typedef __attribute__((ext_vector_type(16))) __bf16 v16bf;
typedef __attribute__((ext_vector_type(8)))  float  v8f;

#define RVQ_B    16
#define RVQ_D    256
#define RVQ_T    2048
#define RVQ_K    1024
#define RVQ_NCB  8
#define RVQ_N    (RVQ_B * RVQ_T)          // 32768 rows
#define OUT_Q    (RVQ_B * RVQ_D * RVQ_T)  // 8388608 floats, then 4 scalars

// workspace layout (bytes)
#define WS_CB    0                        // bf16 codebooks: 8*1024*256*2 = 4 MiB
#define WS_EN    4194304                  // fp32 0.5*||e||^2 : 8192*4
#define WS_FC    4227072                  // int frame_counts : 32768*4
#define WS_OCC   4358144                  // u32 occupancy flags : 8192*4
#define WS_ST    4390912                  // fp32 stats: [0..7]=sse, [8..15]=cnt, [16]=sum_fc

// ---------------------------------------------------------------------------
__global__ void k_zero(unsigned* __restrict__ occ, float* __restrict__ st) {
    int tid = threadIdx.x;
    for (int i = tid; i < RVQ_NCB * RVQ_K; i += 256) occ[i] = 0u;
    if (tid < 32) st[tid] = 0.f;
}

// One block per (cb,k): quantize code to bf16, compute 0.5*||e_q||^2.
__global__ __launch_bounds__(256) void k_convert(const float* __restrict__ embeds,
                                                 __bf16* __restrict__ cbq,
                                                 float* __restrict__ en) {
    int idx = blockIdx.x;                 // cb*1024 + k
    int tid = threadIdx.x;
    float v = embeds[(size_t)idx * RVQ_D + tid];
    __bf16 h = (__bf16)v;
    cbq[(size_t)idx * RVQ_D + tid] = h;
    float hv = (float)h;
    float p = hv * hv;
    #pragma unroll
    for (int m = 16; m > 0; m >>= 1) p += __shfl_xor(p, m, 32);
    __shared__ float w8[8];
    if ((tid & 31) == 0) w8[tid >> 5] = p;
    __syncthreads();
    if (tid == 0) {
        float s = 0.f;
        #pragma unroll
        for (int i = 0; i < 8; ++i) s += w8[i];
        en[idx] = 0.5f * s;
    }
}

// One block per batch b: energy, per-batch min/max, complexity, frame_counts,
// plus global accumulation of sum_fc and per-cb mask counts.
__global__ __launch_bounds__(256) void k_energy(const float* __restrict__ x,
                                                const float* __restrict__ thr,
                                                int* __restrict__ fc,
                                                float* __restrict__ st) {
    __shared__ float se[RVQ_T];
    __shared__ float smn[256], smx[256];
    int b = blockIdx.x, tid = threadIdx.x;
    const float* xb = x + (size_t)b * RVQ_D * RVQ_T;
    #pragma unroll 1
    for (int tt = 0; tt < 8; ++tt) {
        int t = tt * 256 + tid;
        float s = 0.f;
        for (int d = 0; d < RVQ_D; ++d) {
            float v = xb[(size_t)d * RVQ_T + t];   // coalesced across tid
            s += v * v;
        }
        se[t] = s * (1.f / RVQ_D);
    }
    __syncthreads();
    float mn = 3.4e38f, mx = -3.4e38f;
    for (int i = tid; i < RVQ_T; i += 256) { mn = fminf(mn, se[i]); mx = fmaxf(mx, se[i]); }
    smn[tid] = mn; smx[tid] = mx;
    __syncthreads();
    for (int s = 128; s > 0; s >>= 1) {
        if (tid < s) { smn[tid] = fminf(smn[tid], smn[tid + s]); smx[tid] = fmaxf(smx[tid], smx[tid + s]); }
        __syncthreads();
    }
    mn = smn[0]; mx = smx[0];
    float inv = 1.f / fmaxf(mx - mn, 1e-6f);
    float th[7];
    #pragma unroll
    for (int i = 0; i < 7; ++i) th[i] = thr[i];
    float lsum = 0.f;
    float lcnt[RVQ_NCB];
    #pragma unroll
    for (int cb = 0; cb < RVQ_NCB; ++cb) lcnt[cb] = 0.f;
    #pragma unroll 1
    for (int tt = 0; tt < 8; ++tt) {
        int t = tt * 256 + tid;
        float cx = (se[t] - mn) * inv;
        int f = 1;
        #pragma unroll
        for (int i = 0; i < 7; ++i) f += (cx > th[i]) ? 1 : 0;
        fc[b * RVQ_T + t] = f;
        lsum += (float)f;
        #pragma unroll
        for (int cb = 0; cb < RVQ_NCB; ++cb) lcnt[cb] += (f > cb) ? 1.f : 0.f;
    }
    #pragma unroll
    for (int m = 16; m > 0; m >>= 1) {
        lsum += __shfl_xor(lsum, m, 32);
        #pragma unroll
        for (int cb = 0; cb < RVQ_NCB; ++cb) lcnt[cb] += __shfl_xor(lcnt[cb], m, 32);
    }
    if ((tid & 31) == 0) {
        atomicAdd(&st[16], lsum);
        #pragma unroll
        for (int cb = 0; cb < RVQ_NCB; ++cb) atomicAdd(&st[8 + cb], lcnt[cb]);
    }
}

// ---------------------------------------------------------------------------
// Main fused kernel: 512 WGs x 128 threads (4 waves), 16 rows per wave.
// Residual held in registers as bf16 hi/lo pair in WMMA A-fragment layout:
//   lane L: row = L&15,  element (c,e): d = 32c + 16*(e>>3) + 8*(L>>4) + (e&7)
// ---------------------------------------------------------------------------
__global__ __launch_bounds__(128, 1) void k_main(const float* __restrict__ x,
                                                 const float* __restrict__ embeds,
                                                 const __bf16* __restrict__ cbq,
                                                 const float* __restrict__ en,
                                                 const int* __restrict__ fc,
                                                 unsigned* __restrict__ occ,
                                                 float* __restrict__ st,
                                                 float* __restrict__ out) {
    // double-buffered codebook chunk (64 codes x 256 dims, bf16) + enorms
    __shared__ __align__(32) __bf16 s_cb[2][64 * RVQ_D];   // 2 x 32 KiB
    __shared__ float s_en[2][64];

    const int tid  = threadIdx.x;
    const int lane = tid & 31;
    const int hl   = lane >> 4;
    const int lm   = lane & 15;
    const int wave = tid >> 5;
    const int r    = blockIdx.x * 64 + wave * 16 + lm;  // flat row = b*T + t
    const int b    = r >> 11;
    const int t    = r & (RVQ_T - 1);
    const float* xbase = x   + (size_t)b * RVQ_D * RVQ_T + t;
    float*       obase = out + (size_t)b * RVQ_D * RVQ_T + t;
    const int fcv = fc[r];

    // ---- load x into hi/lo bf16 residual (A-fragment layout) ----
    v16bf ahi[8], alo[8];
    #pragma unroll
    for (int c = 0; c < 8; ++c) {
        v16bf hi, lo;
        #pragma unroll
        for (int e = 0; e < 16; ++e) {
            int d = 32 * c + 16 * (e >> 3) + 8 * hl + (e & 7);
            float v = xbase[(size_t)d * RVQ_T];
            __bf16 h = (__bf16)v;
            hi[e] = h;
            lo[e] = (__bf16)(v - (float)h);
        }
        ahi[c] = hi; alo[c] = lo;
    }

    #pragma unroll 1
    for (int cb = 0; cb < RVQ_NCB; ++cb) {
        float bestV[8]; int bestI[8];
        #pragma unroll
        for (int v = 0; v < 8; ++v) { bestV[v] = -3.4e38f; bestI[v] = 0; }

        const __bf16* cbg = cbq + (size_t)cb * RVQ_K * RVQ_D;
        const float*  eng = en + cb * RVQ_K;

        // async-stage one 64-code chunk (32 KiB) into buffer `buf`
        auto stage = [&](int chunk, int buf) {
            const __bf16* gsrc = cbg + (size_t)chunk * 64 * RVQ_D;
            unsigned lbase = (unsigned)(uintptr_t)(&s_cb[buf][0]);
            #pragma unroll
            for (int i = 0; i < 16; ++i) {
                unsigned off = (unsigned)((tid + i * 128) * 16);
                asm volatile("global_load_async_to_lds_b128 %0, %1, %2"
                             :: "v"(lbase + off), "v"(off), "s"(gsrc)
                             : "memory");
            }
            if (tid < 64) s_en[buf][tid] = eng[chunk * 64 + tid];
        };

        // prologue: fill buffer 0 with chunk 0
        stage(0, 0);
        asm volatile("s_wait_asynccnt 0x0" ::: "memory");
        __syncthreads();

        #pragma unroll 1
        for (int ch = 0; ch < 16; ++ch) {           // 64 codes per chunk
            const int cur = ch & 1;
            if (ch < 15) stage(ch + 1, cur ^ 1);    // overlap copy with compute

            // two independent accumulator chains per chunk (breaks WMMA RAW chain)
            #pragma unroll
            for (int tlp = 0; tlp < 2; ++tlp) {
                int n0 = ch * 64 + tlp * 32;
                v8f acc0 = {0.f, 0.f, 0.f, 0.f, 0.f, 0.f, 0.f, 0.f};
                v8f acc1 = {0.f, 0.f, 0.f, 0.f, 0.f, 0.f, 0.f, 0.f};
                const __bf16* brow0 = &s_cb[cur][(tlp * 32 + lm) * RVQ_D + hl * 8];
                const __bf16* brow1 = brow0 + 16 * RVQ_D;
                #pragma unroll
                for (int kc = 0; kc < 8; ++kc) {
                    typedef __attribute__((ext_vector_type(8))) __bf16 v8bf;
                    v8bf b0l = *(const v8bf*)(brow0 + kc * 32);
                    v8bf b0h = *(const v8bf*)(brow0 + kc * 32 + 16);
                    v8bf b1l = *(const v8bf*)(brow1 + kc * 32);
                    v8bf b1h = *(const v8bf*)(brow1 + kc * 32 + 16);
                    v16bf b0 = __builtin_shufflevector(b0l, b0h, 0,1,2,3,4,5,6,7,8,9,10,11,12,13,14,15);
                    v16bf b1 = __builtin_shufflevector(b1l, b1h, 0,1,2,3,4,5,6,7,8,9,10,11,12,13,14,15);
                    acc0 = __builtin_amdgcn_wmma_f32_16x16x32_bf16(
                        false, ahi[kc], false, b0, (short)0, acc0, false, false);
                    acc1 = __builtin_amdgcn_wmma_f32_16x16x32_bf16(
                        false, ahi[kc], false, b1, (short)0, acc1, false, false);
                    acc0 = __builtin_amdgcn_wmma_f32_16x16x32_bf16(
                        false, alo[kc], false, b0, (short)0, acc0, false, false);
                    acc1 = __builtin_amdgcn_wmma_f32_16x16x32_bf16(
                        false, alo[kc], false, b1, (short)0, acc1, false, false);
                }
                float en0 = s_en[cur][tlp * 32 + lm];
                float en1 = s_en[cur][tlp * 32 + 16 + lm];
                #pragma unroll
                for (int v = 0; v < 8; ++v) {
                    float s0 = acc0[v] - en0;      // argmax(s) == argmin d2
                    if (s0 > bestV[v]) { bestV[v] = s0; bestI[v] = n0 + lm; }
                    float s1 = acc1[v] - en1;
                    if (s1 > bestV[v]) { bestV[v] = s1; bestI[v] = n0 + 16 + lm; }
                }
            }
            // next buffer's async copies must have landed before next iteration
            asm volatile("s_wait_asynccnt 0x0" ::: "memory");
            __syncthreads();
        }

        // ---- argmin reduce across the 16 column-lanes of each half-wave ----
        #pragma unroll
        for (int v = 0; v < 8; ++v) {
            #pragma unroll
            for (int m = 1; m < 16; m <<= 1) {
                float ov = __shfl_xor(bestV[v], m, 32);
                int   oi = __shfl_xor(bestI[v], m, 32);
                if (ov > bestV[v] || (ov == bestV[v] && oi < bestI[v])) {
                    bestV[v] = ov; bestI[v] = oi;
                }
            }
        }
        // broadcast per-row winning index; lanes 0..15 mark occupancy
        int myIdx = 0;
        #pragma unroll
        for (int m = 0; m < 16; ++m) {
            int iv = __shfl(bestI[m & 7], (m < 8) ? 0 : 16, 32);
            if (lm == m) myIdx = iv;
            if (lane == m) occ[cb * RVQ_K + iv] = 1u;
        }

        // ---- gather exact fp32 code, commit SSE, masked residual update ----
        const float* eq = embeds + ((size_t)cb * RVQ_K + myIdx) * RVQ_D;
        float sseL = 0.f;
        const bool upd = (fcv > cb);
        #pragma unroll
        for (int c = 0; c < 8; ++c) {
            #pragma unroll
            for (int h2 = 0; h2 < 2; ++h2) {
                int d0 = 32 * c + 16 * h2 + 8 * hl;
                float4 qa = *(const float4*)(eq + d0);
                float4 qb = *(const float4*)(eq + d0 + 4);
                float qv[8] = {qa.x, qa.y, qa.z, qa.w, qb.x, qb.y, qb.z, qb.w};
                #pragma unroll
                for (int i = 0; i < 8; ++i) {
                    int e = h2 * 8 + i;
                    float f   = (float)ahi[c][e] + (float)alo[c][e];
                    float dif = f - qv[i];                 // (r - q), also new residual
                    sseL += dif * dif;
                    if (upd) {
                        __bf16 h = (__bf16)dif;
                        ahi[c][e] = h;
                        alo[c][e] = (__bf16)(dif - (float)h);
                    }
                }
            }
        }
        #pragma unroll
        for (int m = 16; m > 0; m >>= 1) sseL += __shfl_xor(sseL, m, 32);
        if (lane == 0) atomicAdd(&st[cb], sseL);
    }

    // ---- quantized_total = x - residual_final ----
    #pragma unroll
    for (int c = 0; c < 8; ++c) {
        #pragma unroll
        for (int e = 0; e < 16; ++e) {
            int d = 32 * c + 16 * (e >> 3) + 8 * hl + (e & 7);
            float xv = xbase[(size_t)d * RVQ_T];
            obase[(size_t)d * RVQ_T] = xv - ((float)ahi[c][e] + (float)alo[c][e]);
        }
    }
}

// ---------------------------------------------------------------------------
__global__ __launch_bounds__(256) void k_stats(const unsigned* __restrict__ occ,
                                               const float* __restrict__ st,
                                               float* __restrict__ out) {
    __shared__ float red[256];
    int tid = threadIdx.x;
    float used = 0.f;
    for (int i = tid; i < RVQ_NCB * RVQ_K; i += 256) used += (occ[i] != 0u) ? 1.f : 0.f;
    red[tid] = used;
    __syncthreads();
    for (int s = 128; s > 0; s >>= 1) {
        if (tid < s) red[tid] += red[tid + s];
        __syncthreads();
    }
    if (tid == 0) {
        float tc = 0.f, bits = 0.f;
        #pragma unroll
        for (int cb = 0; cb < RVQ_NCB; ++cb) {
            // commit*mean(mask) = (2*SSE/(N*D)) * (cnt/N)
            tc += (2.f * st[cb] / (float)(RVQ_N * (size_t)RVQ_D)) * (st[8 + cb] / (float)RVQ_N);
            bits += st[8 + cb];
        }
        out[OUT_Q + 0] = tc;
        out[OUT_Q + 1] = red[0] / (float)(RVQ_NCB * RVQ_K);  // avg util
        out[OUT_Q + 2] = st[16] / (float)RVQ_N;              // avg frames
        out[OUT_Q + 3] = bits / (float)RVQ_N;                // avg bits
    }
}

// ---------------------------------------------------------------------------
extern "C" void kernel_launch(void* const* d_in, const int* in_sizes, int n_in,
                              void* d_out, int out_size, void* d_ws, size_t ws_size,
                              hipStream_t stream) {
    (void)in_sizes; (void)n_in; (void)out_size; (void)ws_size;
    const float* x      = (const float*)d_in[0];   // [16,256,2048]
    const float* embeds = (const float*)d_in[1];   // [8,1024,256]
    const float* thr    = (const float*)d_in[2];   // [7]
    float* out = (float*)d_out;

    char* w = (char*)d_ws;
    __bf16*   ws_cb  = (__bf16*)(w + WS_CB);
    float*    ws_en  = (float*)(w + WS_EN);
    int*      ws_fc  = (int*)(w + WS_FC);
    unsigned* ws_occ = (unsigned*)(w + WS_OCC);
    float*    ws_st  = (float*)(w + WS_ST);

    k_zero<<<1, 256, 0, stream>>>(ws_occ, ws_st);
    k_convert<<<RVQ_NCB * RVQ_K, 256, 0, stream>>>(embeds, ws_cb, ws_en);
    k_energy<<<RVQ_B, 256, 0, stream>>>(x, thr, ws_fc, ws_st);
    k_main<<<RVQ_N / 64, 128, 0, stream>>>(x, embeds, ws_cb, ws_en, ws_fc, ws_occ, ws_st, out);
    k_stats<<<1, 256, 0, stream>>>(ws_occ, ws_st, out);
}
